// DividedAttention_5970004541508
// MI455X (gfx1250) — compile-verified, matrix operand order checked by reading, not run on hardware
//
#include <hip/hip_runtime.h>
#include <hip/hip_bf16.h>

typedef __bf16 bf16;
typedef __attribute__((ext_vector_type(16))) __bf16 v16bf;
typedef __attribute__((ext_vector_type(8)))  float  v8f;

#define B_    16
#define NSP   196
#define FT    8
#define DIM   512
#define HEADS 8
#define DHEAD 64
#define NTOK  1569                 // 1 + FT*NSP
#define MROWS (B_ * NTOK)          // 25104
#define INNER 512

// ---------------------------------------------------------------------------
// WMMA helpers (CDNA5 wave32, 16x16x32 bf16, f32 accum)
// ---------------------------------------------------------------------------
static __device__ __forceinline__ v8f wmma_bf16(v16bf a, v16bf b, v8f c) {
  return __builtin_amdgcn_wmma_f32_16x16x32_bf16(
      /*neg_a=*/false, a, /*neg_b=*/false, b,
      /*c_mod=*/(short)0, c, /*reuse_a=*/false, /*reuse_b=*/false);
}

// Load a 16x32 bf16 fragment from a tile whose rows are contiguous with
// leading dim `ld` (elements). ISA layout: lane<16 holds row (lane&15) with
// K = {h*8 .. h*8+7, 16+h*8 .. 16+h*8+7}, h = lane>>4.
static __device__ __forceinline__ v16bf frag_ld(const bf16* p, int ld, int lane) {
  const int r = lane & 15;
  const int h = (lane >> 4) & 1;
  const bf16* q = p + r * ld + h * 8;
  v16bf f;
#pragma unroll
  for (int i = 0; i < 8; ++i) { f[i] = q[i]; f[i + 8] = q[i + 16]; }
  return f;
}

// Async global->LDS copy, 16 bytes per active lane. lds_byte_off is the
// per-lane LDS byte address (relative to the wave's LDS base); gptr is the
// per-lane global address. Tracked with ASYNCcnt.
static __device__ __forceinline__ void async_copy_b128(unsigned lds_byte_off,
                                                       const void* gptr) {
  asm volatile("global_load_async_to_lds_b128 %0, %1, off"
               :: "v"(lds_byte_off), "v"(gptr) : "memory");
}
static __device__ __forceinline__ void wait_async0() {
  asm volatile("s_wait_asynccnt 0" ::: "memory");
}

// ---------------------------------------------------------------------------
// fp32 -> bf16 conversion
// ---------------------------------------------------------------------------
__global__ void cvt_f32_bf16(const float* __restrict__ src, bf16* __restrict__ dst, int n) {
  int i = blockIdx.x * blockDim.x + threadIdx.x;
  if (i < n) dst[i] = (bf16)src[i];
}

// ---------------------------------------------------------------------------
// Tiled bf16 GEMM: C[M,Nc] = A[M,K] * W[K,Nc]
//   mode 0: store fp32 C + bias  (output projection -> d_out)
//   mode 1: scatter bf16 into head-split q/k/v buffers [(b*H+h)][n][d]
// Block tile 128x128x32, 256 threads = 8 waves in 2x4, wave tile 64x32.
// A tile is staged with async global->LDS b128 copies (ASYNCcnt path);
// W tile is transposed through VGPRs so B-fragments stay contiguous in LDS.
// ---------------------------------------------------------------------------
#define BM  128
#define BN  128
#define BK  32
#define LDA 40
#define LDB 40

__global__ __launch_bounds__(256) void gemm_bf16_kernel(
    const bf16* __restrict__ A, const bf16* __restrict__ W,
    int M, int K, int Nc, int mode,
    float* __restrict__ outF, const float* __restrict__ bias,
    bf16* __restrict__ qb, bf16* __restrict__ kb, bf16* __restrict__ vb)
{
  // NOTE: declaration order fixes LDS layout: As at byte 0, Bs at BM*LDA*2.
  __shared__ bf16 As[BM * LDA];
  __shared__ bf16 Bs[BN * LDB];

  const int tid  = threadIdx.x;
  const int lane = tid & 31;
  const int wave = tid >> 5;      // 0..7
  const int wr   = wave >> 2;     // 0..1
  const int wc   = wave & 3;      // 0..3
  const int rowBase = blockIdx.x * BM;
  const int colBase = blockIdx.y * BN;

  v8f zero = {};
  v8f acc[4][2];
#pragma unroll
  for (int i = 0; i < 4; ++i)
#pragma unroll
    for (int j = 0; j < 2; ++j) acc[i][j] = zero;

  const int ar  = tid >> 1;          // 0..127 : row of A tile
  const int ac  = (tid & 1) * 16;    // 0 | 16 : k-half of A tile
  const int bkk = tid & 31;          // k row of W tile
  const int bcg = wave * 16;         // column group base of W tile

  const unsigned aLds = (unsigned)((ar * LDA + ac) * sizeof(bf16)); // As base = 0

  for (int k0 = 0; k0 < K; k0 += BK) {
    // -- stage A tile (row-major [128][32]) via async global->LDS
    {
      const int m = rowBase + ar;
      if (m < M) {
        const bf16* g = A + (size_t)m * K + k0 + ac;
        async_copy_b128(aLds,      (const void*)g);
        async_copy_b128(aLds + 16, (const void*)(g + 8));
        if (k0 + BK < K)   // pull next tile toward caches (global_prefetch_b8)
          __builtin_prefetch((const void*)(g + BK), 0, 0);
      } else {
        uint4 z = make_uint4(0u, 0u, 0u, 0u);
        uint4* s = (uint4*)(As + ar * LDA + ac);
        s[0] = z; s[1] = z;
      }
    }
    // -- stage W tile transposed (Bs[col][k]) so B-fragments are contiguous
    {
      const bf16* g = W + (size_t)(k0 + bkk) * Nc + colBase + bcg;
      union { uint4 u[2]; bf16 e[16]; } buf;
      buf.u[0] = ((const uint4*)g)[0];
      buf.u[1] = ((const uint4*)g)[1];
#pragma unroll
      for (int i = 0; i < 16; ++i) Bs[(bcg + i) * LDB + bkk] = buf.e[i];
    }
    wait_async0();
    __syncthreads();

    v16bf af[4], bfg[2];
#pragma unroll
    for (int i = 0; i < 4; ++i) af[i]  = frag_ld(As + (wr * 64 + i * 16) * LDA, LDA, lane);
#pragma unroll
    for (int j = 0; j < 2; ++j) bfg[j] = frag_ld(Bs + (wc * 32 + j * 16) * LDB, LDB, lane);
#pragma unroll
    for (int i = 0; i < 4; ++i)
#pragma unroll
      for (int j = 0; j < 2; ++j)
        acc[i][j] = wmma_bf16(af[i], bfg[j], acc[i][j]);
    __syncthreads();
  }

  // -- epilogue. C/D layout: row = v + 8*(lane>>4), col = lane&15 in each tile
  const int lr = lane & 15;
  const int lh = (lane >> 4) & 1;
#pragma unroll
  for (int i = 0; i < 4; ++i) {
#pragma unroll
    for (int j = 0; j < 2; ++j) {
#pragma unroll
      for (int v = 0; v < 8; ++v) {
        const int row = rowBase + wr * 64 + i * 16 + v + 8 * lh;
        const int col = colBase + wc * 32 + j * 16 + lr;
        if (row < M) {
          const float val = acc[i][j][v];
          if (mode == 0) {
            outF[(size_t)row * Nc + col] = val + bias[col];
          } else {
            const int bI = row / NTOK;
            const int n  = row - bI * NTOK;
            const int which = col >> 9;      // 0:q 1:k 2:v
            const int inner = col & 511;
            const int h = inner >> 6;
            const int d = inner & 63;
            bf16* dst = (which == 0) ? qb : (which == 1) ? kb : vb;
            dst[((size_t)(bI * HEADS + h) * NTOK + n) * DHEAD + d] = (bf16)val;
          }
        }
      }
    }
  }
}

// ---------------------------------------------------------------------------
// Spatial divided attention: one block per (b*H, f) segment.
// 196 queries x 197 keys (cls + 196) x 64 dim, all WMMA.
// Row tiles: 13 (pad 208); key tiles: 14 (pad 224); 4 waves per block.
// K rows staged with async global->LDS b128 copies (dynamic LDS base = 0).
// ---------------------------------------------------------------------------
#define RT   13
#define CTI  14
#define KP   224
#define LDK  72
#define LDVT 240
#define LDP  240

__global__ __launch_bounds__(128) void attn_spatial_kernel(
    const bf16* __restrict__ qb, const bf16* __restrict__ kb,
    const bf16* __restrict__ vb, bf16* __restrict__ out /* [B][NTOK][INNER] bf16 */)
{
  extern __shared__ bf16 smem[];
  bf16* Ks = smem;                  // [KP][LDK]   row-major keys (padded w/ 0)
  bf16* Vt = Ks + KP * LDK;         // [64][LDVT]  V transposed
  bf16* Pw = Vt + 64 * LDVT;        // 4 x [16][LDP] per-wave P scratch

  const int tid  = threadIdx.x;
  const int lane = tid & 31;
  const int wave = tid >> 5;        // 0..3
  const int bhf  = blockIdx.x;      // 0..1023
  const int bh   = bhf >> 3;
  const int f    = bhf & 7;
  const size_t base = (size_t)bh * NTOK * DHEAD;

  // stage K rows (0 = cls key, 1..196 = segment keys, rest = 0) as
  // 16-byte chunks: 224 rows x 8 chunks, async copies straight into LDS.
  for (int c = tid; c < KP * 8; c += 128) {
    const int j = c >> 3;           // key row
    const int u = c & 7;            // 16B chunk within row
    const unsigned lb = (unsigned)((j * LDK + u * 8) * sizeof(bf16)); // Ks base = 0
    if (j <= NSP) {
      const int n = (j == 0) ? 0 : (1 + f * NSP + (j - 1));
      async_copy_b128(lb, (const void*)(kb + base + (size_t)n * DHEAD + u * 8));
    } else {
      *(uint4*)(Ks + j * LDK + u * 8) = make_uint4(0u, 0u, 0u, 0u);
    }
  }
  // stage V transposed: Vt[d][j]
  for (int j = tid; j < KP; j += 128) {
    if (j <= NSP) {
      const int n = (j == 0) ? 0 : (1 + f * NSP + (j - 1));
      const bf16* g = vb + base + (size_t)n * DHEAD;
#pragma unroll
      for (int d = 0; d < 64; ++d) Vt[d * LDVT + j] = g[d];
    } else {
#pragma unroll
      for (int d = 0; d < 64; ++d) Vt[d * LDVT + j] = (bf16)0.0f;
    }
  }
  wait_async0();
  __syncthreads();

  bf16* P = Pw + wave * 16 * LDP;
  const int r = lane & 15;
  const int h = (lane >> 4) & 1;

  for (int t = wave; t < RT; t += 4) {
    // Q fragments straight from global (rows clamped; pad rows discarded later)
    int qrow = t * 16 + r; if (qrow > NSP - 1) qrow = NSP - 1;
    const bf16* qp = qb + base + (size_t)(1 + f * NSP + qrow) * DHEAD;
    v16bf qf0, qf1;
#pragma unroll
    for (int i = 0; i < 8; ++i) {
      qf0[i] = qp[h * 8 + i];      qf0[i + 8] = qp[16 + h * 8 + i];
      qf1[i] = qp[32 + h * 8 + i]; qf1[i + 8] = qp[48 + h * 8 + i];
    }

    // S = (Q K^T) * scale, masked past key 197
    v8f S[CTI];
#pragma unroll
    for (int ct = 0; ct < CTI; ++ct) {
      v8f s = {};
      v16bf kf0 = frag_ld(Ks + ct * 16 * LDK, LDK, lane);
      v16bf kf1 = frag_ld(Ks + ct * 16 * LDK + 32, LDK, lane);
      s = wmma_bf16(qf0, kf0, s);
      s = wmma_bf16(qf1, kf1, s);
      const float m = (ct * 16 + r < NSP + 1) ? 0.0f : -1e30f;
#pragma unroll
      for (int v = 0; v < 8; ++v) s[v] = s[v] * 0.125f + m;
      S[ct] = s;
    }

    // row softmax: each row lives in one 16-lane half (xor-shuffle reduce)
#pragma unroll
    for (int v = 0; v < 8; ++v) {
      float mx = -1e30f;
#pragma unroll
      for (int ct = 0; ct < CTI; ++ct) mx = fmaxf(mx, S[ct][v]);
#pragma unroll
      for (int off = 1; off < 16; off <<= 1) mx = fmaxf(mx, __shfl_xor(mx, off, 32));
      float su = 0.0f;
#pragma unroll
      for (int ct = 0; ct < CTI; ++ct) { float e = __expf(S[ct][v] - mx); S[ct][v] = e; su += e; }
#pragma unroll
      for (int off = 1; off < 16; off <<= 1) su += __shfl_xor(su, off, 32);
      const float inv = 1.0f / su;
#pragma unroll
      for (int ct = 0; ct < CTI; ++ct)
        P[(v + 8 * h) * LDP + ct * 16 + r] = (bf16)(S[ct][v] * inv);
    }

    // O = P * V  (per-wave LDS is private; same-wave DS ops are in-order)
    const int bI = bh >> 3, hh = bh & 7;
#pragma unroll
    for (int oc = 0; oc < 4; ++oc) {
      v8f o = {};
#pragma unroll
      for (int ks = 0; ks < 7; ++ks) {
        v16bf pf = frag_ld(P + ks * 32, LDP, lane);
        v16bf vf = frag_ld(Vt + oc * 16 * LDVT + ks * 32, LDVT, lane);
        o = wmma_bf16(pf, vf, o);
      }
#pragma unroll
      for (int v = 0; v < 8; ++v) {
        const int rr = t * 16 + v + 8 * h;
        if (rr < NSP) {
          const int n2 = 1 + f * NSP + rr;
          out[((size_t)bI * NTOK + n2) * INNER + hh * DHEAD + oc * 16 + r] = (bf16)o[v];
        }
      }
    }
  }
}

// ---------------------------------------------------------------------------
// CLS attention: one block per (b*H); 1 query x 1569 keys (tiny, scalar)
// ---------------------------------------------------------------------------
__global__ __launch_bounds__(256) void attn_cls_kernel(
    const bf16* __restrict__ qb, const bf16* __restrict__ kb,
    const bf16* __restrict__ vb, bf16* __restrict__ out)
{
  __shared__ float logits[NTOK];
  __shared__ float red[256];
  __shared__ float qs[64];
  __shared__ float partial[4][64];

  const int tid = threadIdx.x;
  const int bh  = blockIdx.x;
  const size_t base = (size_t)bh * NTOK * DHEAD;

  if (tid < 64) qs[tid] = (float)qb[base + tid];
  __syncthreads();

  float lmax = -1e30f;
  for (int j = tid; j < NTOK; j += 256) {
    const bf16* kp = kb + base + (size_t)j * DHEAD;
    float dot = 0.0f;
#pragma unroll
    for (int d = 0; d < 64; ++d) dot += qs[d] * (float)kp[d];
    dot *= 0.125f;
    logits[j] = dot;
    lmax = fmaxf(lmax, dot);
  }
  red[tid] = lmax; __syncthreads();
  for (int s = 128; s > 0; s >>= 1) {
    if (tid < s) red[tid] = fmaxf(red[tid], red[tid + s]);
    __syncthreads();
  }
  const float m = red[0];
  __syncthreads();

  float lsum = 0.0f;
  for (int j = tid; j < NTOK; j += 256) {
    float e = __expf(logits[j] - m);
    logits[j] = e; lsum += e;
  }
  red[tid] = lsum; __syncthreads();
  for (int s = 128; s > 0; s >>= 1) {
    if (tid < s) red[tid] += red[tid + s];
    __syncthreads();
  }
  const float inv = 1.0f / red[0];

  const int d = tid & 63, ch = tid >> 6;
  float o = 0.0f;
  for (int j = ch; j < NTOK; j += 4)
    o += logits[j] * (float)vb[base + (size_t)j * DHEAD + d];
  partial[ch][d] = o; __syncthreads();
  if (tid < 64) {
    const float rres = (partial[0][tid] + partial[1][tid] +
                        partial[2][tid] + partial[3][tid]) * inv;
    const int bI = bh >> 3, hh = bh & 7;
    out[((size_t)bI * NTOK + 0) * INNER + hh * DHEAD + tid] = (bf16)rres;
  }
}

// ---------------------------------------------------------------------------
// launch
// ---------------------------------------------------------------------------
extern "C" void kernel_launch(void* const* d_in, const int* in_sizes, int n_in,
                              void* d_out, int out_size, void* d_ws, size_t ws_size,
                              hipStream_t stream) {
  const float* x    = (const float*)d_in[0];
  // d_in[1] = f (scalar, fixed at FT)
  const float* Wqkv = (const float*)d_in[2];
  const float* Wout = (const float*)d_in[3];
  const float* bout = (const float*)d_in[4];
  float* outp = (float*)d_out;

  char* ws = (char*)d_ws;
  auto alloc = [&](size_t bytes) -> char* {
    char* p = ws; ws += (bytes + 255) & ~(size_t)255; return p;
  };
  bf16* xb    = (bf16*)alloc((size_t)MROWS * DIM * 2);
  bf16* wqkvb = (bf16*)alloc((size_t)DIM * 3 * INNER * 2);
  bf16* woutb = (bf16*)alloc((size_t)INNER * DIM * 2);
  bf16* qbuf  = (bf16*)alloc((size_t)B_ * HEADS * NTOK * DHEAD * 2);
  bf16* kbuf  = (bf16*)alloc((size_t)B_ * HEADS * NTOK * DHEAD * 2);
  bf16* vbuf  = (bf16*)alloc((size_t)B_ * HEADS * NTOK * DHEAD * 2);
  bf16* abuf  = (bf16*)alloc((size_t)MROWS * INNER * 2);

  const int nx = MROWS * DIM;
  cvt_f32_bf16<<<(nx + 255) / 256, 256, 0, stream>>>(x, xb, nx);
  cvt_f32_bf16<<<(DIM * 3 * INNER + 255) / 256, 256, 0, stream>>>(Wqkv, wqkvb, DIM * 3 * INNER);
  cvt_f32_bf16<<<(INNER * DIM + 255) / 256, 256, 0, stream>>>(Wout, woutb, INNER * DIM);

  // QKV projection: 25104 x 512 x 1536 -> head-split bf16 q/k/v
  {
    dim3 grid((MROWS + BM - 1) / BM, (3 * INNER) / BN);
    gemm_bf16_kernel<<<grid, 256, 0, stream>>>(
        xb, wqkvb, MROWS, DIM, 3 * INNER, /*mode=*/1,
        nullptr, nullptr, qbuf, kbuf, vbuf);
  }

  // Divided attention
  {
    const size_t smem = (size_t)(KP * LDK + 64 * LDVT + 4 * 16 * LDP) * sizeof(bf16);
    attn_spatial_kernel<<<B_ * HEADS * FT, 128, smem, stream>>>(qbuf, kbuf, vbuf, abuf);
    attn_cls_kernel<<<B_ * HEADS, 256, 0, stream>>>(qbuf, kbuf, vbuf, abuf);
  }

  // Output projection + bias: 25104 x 512 x 512 -> fp32 d_out
  {
    dim3 grid((MROWS + BM - 1) / BM, INNER / BN);
    gemm_bf16_kernel<<<grid, 256, 0, stream>>>(
        abuf, woutb, MROWS, INNER, INNER, /*mode=*/0,
        outp, bout, nullptr, nullptr, nullptr);
  }
}
